// GraphGRU_62895501083193
// MI455X (gfx1250) — matrix operand order repaired and checked by reading, not compiled, not run on hardware
//
#include <hip/hip_runtime.h>
#include <hip/hip_bf16.h>

// ---------------------------------------------------------------------------
// GraphGRU (2-layer GAT-GRU) for MI455X / gfx1250, wave32.
// GEMM path: v_wmma_f32_16x16x32_bf16; W staged to LDS via async DMA
// (global_load_async_to_lds_b128 + s_wait_asynccnt), B frags double-buffered.
// ---------------------------------------------------------------------------

#define T_STEPS 16
#define B_GR    8
#define N_NODES 5000
#define DIN     64
#define HID     64
#define F3      192          // 3*HID
#define KDIM    128          // DIN+HID == HID+HID
#define NODES   (B_GR * N_NODES)   // 40000
#define W_FRAG_USHORTS 24576       // 128*192 bf16 per layer

typedef __attribute__((ext_vector_type(16))) __bf16 v16bf;
typedef __attribute__((ext_vector_type(8)))  float  v8f;

union Frag { uint4 u[2]; v16bf v; };

__device__ __forceinline__ unsigned short f2bf(float f) {
    unsigned u = __float_as_uint(f);
    u = (u + 0x7FFFu + ((u >> 16) & 1u)) >> 16;     // round-to-nearest-even
    return (unsigned short)u;
}
__device__ __forceinline__ unsigned mapf(float f) {  // order-preserving uint
    unsigned u = __float_as_uint(f);
    return (u & 0x80000000u) ? ~u : (u | 0x80000000u);
}
__device__ __forceinline__ float unmapf(unsigned u) {
    u = (u & 0x80000000u) ? (u & 0x7FFFFFFFu) : ~u;
    return __uint_as_float(u);
}
__device__ __forceinline__ bool edge_decode(const int* __restrict__ ei, int E,
                                            int eidx, int& s, int& d) {
    if (eidx < E) { s = ei[eidx]; d = ei[E + eidx]; return s != d; }
    s = d = eidx - E;                // appended self-loop
    return true;
}

// --- W (128x192 f32) -> bf16, pre-swizzled into per-lane WMMA B fragments ---
// frag block fb = kc*12 + j ; within: lane (0..31) * 16 contiguous bf16.
// B layout: lanes 0-15 -> N=lane, K = kc*32 + 0..15 ; lanes 16-31 -> K 16..31.
__global__ void convert_w_kernel(const float* __restrict__ W0,
                                 const float* __restrict__ W1,
                                 unsigned short* __restrict__ Wb) {
    int tid = blockIdx.x * blockDim.x + threadIdx.x;
    if (tid >= 2 * W_FRAG_USHORTS) return;
    int mat  = tid / W_FRAG_USHORTS;
    int r    = tid % W_FRAG_USHORTS;
    int fb   = r >> 9;              // /512 (32 lanes * 16 elems)
    int lane = (r >> 4) & 31;
    int e    = r & 15;
    int kc = fb / 12, j = fb % 12;
    int k = kc * 32 + ((lane >= 16) ? 16 : 0) + e;
    int n = j * 16 + (lane & 15);
    const float* W = mat ? W1 : W0;
    Wb[tid] = f2bf(W[k * F3 + n]);
}

__global__ void zero_h_kernel(float* __restrict__ h) {
    int tid = blockIdx.x * blockDim.x + threadIdx.x;
    if (tid < NODES * HID) h[tid] = 0.0f;
}

// --- A = [x_t | h] as bf16, row-major (NODES x 128) ---
__global__ void pack_kernel(const float* __restrict__ xin,
                            const float* __restrict__ h,
                            unsigned short* __restrict__ A) {
    int tid = blockIdx.x * blockDim.x + threadIdx.x;
    if (tid >= NODES * KDIM) return;
    int row = tid >> 7, k = tid & 127;
    float v = (k < 64) ? xin[row * 64 + k] : h[row * 64 + (k - 64)];
    A[tid] = f2bf(v);
}

__device__ __forceinline__ void bload(Frag& f, const unsigned short* lw,
                                      int st, int lane) {
    const uint4* p = (const uint4*)(lw + st * 512 + lane * 16);
    f.u[0] = p[0];
    f.u[1] = p[1];
}

// --- GEMM: hfeat(40000x192) = A(40000x128,bf16) @ W(128x192,bf16), f32 acc ---
// 256 threads = 8 waves; each wave: 16 rows x 192 cols (12 accumulators).
__global__ void __launch_bounds__(256)
gemm_kernel(const unsigned short* __restrict__ A,
            const unsigned short* __restrict__ Wb,   // layer's fragment buffer
            float* __restrict__ hfeat) {
    __shared__ unsigned short lw[W_FRAG_USHORTS];    // 48 KB

    // Stage W into LDS with async DMA (memory -> LDS, no VGPR round trip).
    // 3072 x b128 transfers, GVS addressing: saddr = Wb, vaddr = byte offset.
    {
        unsigned ldsBase = (unsigned)(size_t)lw;
        for (int i = threadIdx.x; i < 3072; i += 256) {
            unsigned goff  = (unsigned)i * 16u;
            unsigned laddr = ldsBase + (unsigned)i * 16u;
            asm volatile("global_load_async_to_lds_b128 %0, %1, %2"
                         :: "v"(laddr), "v"(goff), "s"(Wb)
                         : "memory");
        }
        asm volatile("s_wait_asynccnt 0x0" ::: "memory");
    }
    __syncthreads();

    const int wave = threadIdx.x >> 5;
    const int lane = threadIdx.x & 31;
    const int rowBase = blockIdx.x * 128 + wave * 16;
    if (rowBase >= NODES) return;

    const int laneLo = lane & 15;
    const int hiSel  = lane >> 4;                    // 0 or 1

    // Hoist all 4 A fragments (K = 128 in 4 chunks of 32).
    // A frag: lane<16 -> K = kc*32 + 0..15; lane>=16 -> K = kc*32 + 16..31
    const unsigned short* arow = A + (rowBase + laneLo) * KDIM;
    Frag af[4];
    #pragma unroll
    for (int kc = 0; kc < 4; ++kc) {
        const int kbase = kc * 32 + hiSel * 16;
        af[kc].u[0] = *(const uint4*)(arow + kbase);
        af[kc].u[1] = *(const uint4*)(arow + kbase + 8);
    }

    v8f acc[12] = {};
    Frag bf[2];
    bload(bf[0], lw, 0, lane);

    // 48 steps, kc-major; double-buffered B fragments so ds_load of step st+1
    // overlaps the WMMA of step st.
    #pragma unroll
    for (int st = 0; st < 48; ++st) {
        if (st + 1 < 48) bload(bf[(st + 1) & 1], lw, st + 1, lane);
        const int kc = st / 12, j = st % 12;
        acc[j] = __builtin_amdgcn_wmma_f32_16x16x32_bf16(
            false, af[kc].v, false, bf[st & 1].v, (short)0, acc[j], false, false);
    }

    // C/D layout: VGPR v, lanes 0-15 -> M=v, lanes 16-31 -> M=v+8, N=lane&15
    const int rOff = hiSel * 8;
    #pragma unroll
    for (int j = 0; j < 12; ++j) {
        #pragma unroll
        for (int v = 0; v < 8; ++v) {
            hfeat[(rowBase + v + rOff) * F3 + j * 16 + laneLo] = acc[j][v];
        }
    }
}

// --- per-node attention logits: a_s = h . att_s ; a_d = h . att_d ---
__global__ void att_kernel(const float* __restrict__ hfeat,
                           const float* __restrict__ att_s,
                           const float* __restrict__ att_d,
                           float* __restrict__ a_s, float* __restrict__ a_d) {
    int wid  = (blockIdx.x * blockDim.x + threadIdx.x) >> 5;
    int lane = threadIdx.x & 31;
    if (wid >= NODES) return;
    const float* row = hfeat + (size_t)wid * F3;
    float s = 0.f, d = 0.f;
    #pragma unroll
    for (int i = 0; i < 6; ++i) {
        int c = lane + i * 32;
        float hv = row[c];
        s += hv * att_s[c];
        d += hv * att_d[c];
    }
    #pragma unroll
    for (int off = 16; off; off >>= 1) {
        s += __shfl_xor(s, off, 32);
        d += __shfl_xor(d, off, 32);
    }
    if (lane == 0) { a_s[wid] = s; a_d[wid] = d; }
}

__global__ void init_kernel(float* __restrict__ accum,
                            unsigned* __restrict__ m, float* __restrict__ denom) {
    int tid = blockIdx.x * blockDim.x + threadIdx.x;
    if (tid < NODES * F3) accum[tid] = 0.0f;
    if (tid < NODES) { m[tid] = 0x007FFFFFu /* mapf(-inf) */; denom[tid] = 0.0f; }
}

__global__ void edge_max_kernel(const int* __restrict__ ei, int E, int etot,
                                const float* __restrict__ a_s,
                                const float* __restrict__ a_d,
                                unsigned* __restrict__ m) {
    int tid = blockIdx.x * blockDim.x + threadIdx.x;
    if (tid >= etot * B_GR) return;
    int eidx = tid % etot, b = tid / etot;
    int s, d;
    if (!edge_decode(ei, E, eidx, s, d)) return;
    float av = a_s[b * N_NODES + s] + a_d[b * N_NODES + d];
    float e = av > 0.f ? av : 0.2f * av;             // leaky_relu(0.2)
    atomicMax(&m[b * N_NODES + d], mapf(e));
}

__global__ void edge_sum_kernel(const int* __restrict__ ei, int E, int etot,
                                const float* __restrict__ a_s,
                                const float* __restrict__ a_d,
                                const unsigned* __restrict__ m,
                                float* __restrict__ denom) {
    int tid = blockIdx.x * blockDim.x + threadIdx.x;
    if (tid >= etot * B_GR) return;
    int eidx = tid % etot, b = tid / etot;
    int s, d;
    if (!edge_decode(ei, E, eidx, s, d)) return;
    int drow = b * N_NODES + d;
    float av = a_s[b * N_NODES + s] + a_d[drow];
    float e = av > 0.f ? av : 0.2f * av;
    atomicAdd(&denom[drow], __expf(e - unmapf(m[drow])));
}

// one wave per (edge, graph): scatter alpha * h_src into accum[dst]
__global__ void __launch_bounds__(256)
edge_aggr_kernel(const int* __restrict__ ei, int E, int etot,
                 const float* __restrict__ a_s, const float* __restrict__ a_d,
                 const unsigned* __restrict__ m, const float* __restrict__ denom,
                 const float* __restrict__ hfeat, float* __restrict__ accum) {
    int wid  = (blockIdx.x * blockDim.x + threadIdx.x) >> 5;
    int lane = threadIdx.x & 31;
    if (wid >= etot * B_GR) return;
    int eidx = wid % etot, b = wid / etot;
    int s, d;
    if (!edge_decode(ei, E, eidx, s, d)) return;
    int srow = b * N_NODES + s, drow = b * N_NODES + d;
    float av = a_s[srow] + a_d[drow];
    float e = av > 0.f ? av : 0.2f * av;
    float w = __expf(e - unmapf(m[drow])) / denom[drow];
    const float* hs = hfeat + (size_t)srow * F3;
    float* oa = accum + (size_t)drow * F3;
    __builtin_prefetch(hs + lane, 0, 0);             // global_prefetch_b8
    #pragma unroll
    for (int i = 0; i < 6; ++i) {
        int c = lane + i * 32;
        atomicAdd(&oa[c], w * hs[c]);
    }
}

// GRU update: z = sigmoid(conv_z), n = tanh(conv_n); h' = (1-z)*n + z*h
__global__ void gru_update_kernel(const float* __restrict__ accum,
                                  const float* __restrict__ bias,
                                  float* __restrict__ h,
                                  float* __restrict__ out) {
    int tid = blockIdx.x * blockDim.x + threadIdx.x;
    if (tid >= NODES * HID) return;
    int row = tid >> 6, c = tid & 63;
    const float* conv = accum + (size_t)row * F3;
    float zi = conv[64 + c]  + bias[64 + c];
    float ni = conv[128 + c] + bias[128 + c];
    float z  = 1.0f / (1.0f + __expf(-zi));
    float nn = tanhf(ni);
    float hn = (1.0f - z) * nn + z * h[tid];
    h[tid]  = hn;
    out[tid] = hn;
}

extern "C" void kernel_launch(void* const* d_in, const int* in_sizes, int n_in,
                              void* d_out, int out_size, void* d_ws, size_t ws_size,
                              hipStream_t stream) {
    const float* x   = (const float*)d_in[0];
    const int*   ei  = (const int*)d_in[1];
    const float* W0  = (const float*)d_in[2];
    const float* as0 = (const float*)d_in[3];
    const float* ad0 = (const float*)d_in[4];
    const float* b0  = (const float*)d_in[5];
    const float* W1  = (const float*)d_in[6];
    const float* as1 = (const float*)d_in[7];
    const float* ad1 = (const float*)d_in[8];
    const float* b1  = (const float*)d_in[9];
    float* out = (float*)d_out;
    (void)n_in; (void)out_size; (void)ws_size;

    const int E    = in_sizes[1] / 2;     // 80000
    const int etot = E + N_NODES;         // edges + self-loops

    // workspace carve-out (~83 MB total)
    char* ws = (char*)d_ws;
    size_t off = 0;
    auto carve = [&](size_t bytes) -> char* {
        char* p = ws + off;
        off = (off + bytes + 255) & ~(size_t)255;
        return p;
    };
    unsigned short* Wb   = (unsigned short*)carve(2 * W_FRAG_USHORTS * sizeof(unsigned short));
    unsigned short* Abf  = (unsigned short*)carve((size_t)NODES * KDIM * sizeof(unsigned short));
    float*    h     = (float*)carve((size_t)NODES * HID * sizeof(float));
    float*    hfeat = (float*)carve((size_t)NODES * F3 * sizeof(float));
    float*    a_s   = (float*)carve((size_t)NODES * sizeof(float));
    float*    a_d   = (float*)carve((size_t)NODES * sizeof(float));
    unsigned* mm    = (unsigned*)carve((size_t)NODES * sizeof(unsigned));
    float*    denom = (float*)carve((size_t)NODES * sizeof(float));
    float*    accum = (float*)carve((size_t)NODES * F3 * sizeof(float));

    convert_w_kernel<<<(2 * W_FRAG_USHORTS + 255) / 256, 256, 0, stream>>>(W0, W1, Wb);

    const int tot = etot * B_GR;
    for (int layer = 0; layer < 2; ++layer) {
        const float* att_s = layer ? as1 : as0;
        const float* att_d = layer ? ad1 : ad0;
        const float* bias  = layer ? b1  : b0;
        const unsigned short* WbL = Wb + layer * W_FRAG_USHORTS;

        zero_h_kernel<<<(NODES * HID + 255) / 256, 256, 0, stream>>>(h);

        for (int t = 0; t < T_STEPS; ++t) {
            const float* xin = layer ? (out + (size_t)t * NODES * HID)      // layer0 output
                                     : (x   + (size_t)t * NODES * DIN);
            pack_kernel<<<(NODES * KDIM + 255) / 256, 256, 0, stream>>>(xin, h, Abf);
            gemm_kernel<<<(NODES + 127) / 128, 256, 0, stream>>>(Abf, WbL, hfeat);
            att_kernel<<<(NODES + 7) / 8, 256, 0, stream>>>(hfeat, att_s, att_d, a_s, a_d);
            init_kernel<<<(NODES * F3 + 255) / 256, 256, 0, stream>>>(accum, mm, denom);
            edge_max_kernel<<<(tot + 255) / 256, 256, 0, stream>>>(ei, E, etot, a_s, a_d, mm);
            edge_sum_kernel<<<(tot + 255) / 256, 256, 0, stream>>>(ei, E, etot, a_s, a_d, mm, denom);
            edge_aggr_kernel<<<((size_t)tot * 32 + 255) / 256, 256, 0, stream>>>(
                ei, E, etot, a_s, a_d, mm, denom, hfeat, accum);
            gru_update_kernel<<<(NODES * HID + 255) / 256, 256, 0, stream>>>(
                accum, bias, h, out + ((size_t)layer * T_STEPS + t) * NODES * HID);
        }
    }
}
